// GraphConvolutionSparse_23244363006204
// MI455X (gfx1250) — matrix operand order repaired and checked by reading, not compiled, not run on hardware
//
#include <hip/hip_runtime.h>
#include <hip/hip_bf16.h>
#include <stdint.h>

#define N_NODES 100000
#define D_IN    256
#define D_OUT   128
#define NNZX    3200000
#define NNZA    3200000

typedef __attribute__((ext_vector_type(16))) __bf16       v16bf;
typedef __attribute__((ext_vector_type(8)))  float        v8f;
typedef __attribute__((ext_vector_type(4)))  unsigned int u32x4;
typedef __attribute__((ext_vector_type(4)))  __bf16       bf4;
typedef __attribute__((ext_vector_type(4)))  float        f32x4;

// GCC-style int4 vector: matches the async-LDS builtin's parameter type
typedef int gi32x4 __attribute__((vector_size(16)));

#define GASQ __attribute__((address_space(1)))
#define LASQ __attribute__((address_space(3)))

#if defined(__HIP_DEVICE_COMPILE__) && __has_builtin(__builtin_amdgcn_global_load_async_to_lds_b128)
#define HAVE_ASYNC_LDS 1
#endif

// ---- workspace layout (bf16 elements) ----
static const size_t XD_ELEMS = (size_t)N_NODES * D_IN;   // densified X, bf16
static const size_t WT_ELEMS = (size_t)D_OUT * D_IN;     // W transposed [n][k], bf16
static const size_t H_ELEMS  = (size_t)N_NODES * D_OUT;  // h = X@W, bf16
static const size_t WT_OFF   = XD_ELEMS;
static const size_t H_OFF    = XD_ELEMS + WT_ELEMS;
static const size_t WS_BYTES = (XD_ELEMS + WT_ELEMS + H_ELEMS) * 2;

__device__ __forceinline__ unsigned hash_u32(unsigned x) {
    x ^= x >> 16; x *= 0x7feb352du;
    x ^= x >> 15; x *= 0x846ca68bu;
    x ^= x >> 16; return x;
}

// 16B memory->LDS copy: async on CDNA5 (no VGPR round trip), fallback otherwise
__device__ __forceinline__ void copy16_to_lds(const __bf16* __restrict__ gsrc,
                                              __bf16* __restrict__ ldst) {
#ifdef HAVE_ASYNC_LDS
    __builtin_amdgcn_global_load_async_to_lds_b128(
        (GASQ gi32x4*)gsrc, (LASQ gi32x4*)ldst, 0, 0);
#else
    *(u32x4*)ldst = *(const u32x4*)gsrc;
#endif
}

__device__ __forceinline__ void async_wait_all() {
#ifdef HAVE_ASYNC_LDS
#if __has_builtin(__builtin_amdgcn_s_wait_asynccnt)
    __builtin_amdgcn_s_wait_asynccnt(0);
#else
    asm volatile("s_wait_asynccnt 0x0" ::: "memory");
#endif
#endif
}

// ---------- zero fill (grid-stride, 16B stores) ----------
__global__ void zero_k(u32x4* __restrict__ p, size_t n) {
    size_t i = (size_t)blockIdx.x * blockDim.x + threadIdx.x;
    size_t stride = (size_t)gridDim.x * blockDim.x;
    u32x4 z = {0u, 0u, 0u, 0u};
    for (; i < n; i += stride) p[i] = z;
}

// ---------- W[k][n] f32 -> Wt[n][k] bf16 ----------
__global__ void prep_w(const float* __restrict__ W, __bf16* __restrict__ Wt) {
    int t = blockIdx.x * 256 + threadIdx.x;
    if (t >= D_IN * D_OUT) return;
    int k = t >> 7;       // t / 128
    int n = t & 127;
    Wt[(size_t)n * D_IN + k] = (__bf16)W[t];
}

// ---------- sparse dropout + densify into bf16 (CAS accumulate) ----------
__global__ void densify_x(const int* __restrict__ xr, const int* __restrict__ xc,
                          const float* __restrict__ xv, __bf16* __restrict__ Xd, int nnz) {
    int i = blockIdx.x * blockDim.x + threadIdx.x;
    if (i >= nnz) return;
    // mask = floor(0.9 + u) : keep with prob 0.9 (deterministic hash RNG)
    unsigned u = hash_u32((unsigned)i * 0x9e3779b9u + 0x85ebca6bu);
    if (u < 429496730u) return;                       // dropped (~10%)
    float v = xv[i] * (1.0f / 0.9f);
    unsigned idx = (unsigned)xr[i] * D_IN + (unsigned)xc[i];
    unsigned* word = (unsigned*)Xd + (idx >> 1);
    bool hi = (idx & 1u) != 0u;
    unsigned old = *word;
    while (true) {
        unsigned assumed = old;
        unsigned short cur = hi ? (unsigned short)(assumed >> 16)
                                : (unsigned short)(assumed & 0xffffu);
        float f = (float)__builtin_bit_cast(__bf16, cur) + v;
        unsigned short nb = __builtin_bit_cast(unsigned short, (__bf16)f);
        unsigned newv = hi ? ((assumed & 0x0000ffffu) | ((unsigned)nb << 16))
                           : ((assumed & 0xffff0000u) | (unsigned)nb);
        old = atomicCAS(word, assumed, newv);
        if (old == assumed) break;
    }
}

// ---------- dense GEMM: h[100000x128] = Xd[100000x256] @ W, bf16 WMMA ----------
// block = 256 threads = 8 waves; block tile = 32 (M) x 128 (N)
// wave w: m_sub = w>>2 (0..1) -> 16-row strip; (w&3) -> 32-col slab (two 16x16 tiles)
// A-tiles double-buffered in LDS, filled with async global->LDS copies.
__global__ __launch_bounds__(256) void gemm_xw(const __bf16* __restrict__ Xd,
                                               const __bf16* __restrict__ Wt,
                                               __bf16* __restrict__ H) {
    __shared__ __attribute__((aligned(16))) __bf16 As[2][32 * 32];  // 2 x (32 rows x 32 k)
    const int tid   = threadIdx.x;
    const int lane  = tid & 31;
    const int wave  = tid >> 5;
    const int l15   = lane & 15;
    const int lhi   = lane >> 4;
    const int basem = blockIdx.x * 32;
    const int m_sub = wave >> 2;
    const int n0    = (wave & 3) * 32;

    v8f acc0 = {0.f, 0.f, 0.f, 0.f, 0.f, 0.f, 0.f, 0.f};
    v8f acc1 = {0.f, 0.f, 0.f, 0.f, 0.f, 0.f, 0.f, 0.f};

    // staging role: first 128 threads copy one 16B chunk of the 32x32 bf16 tile
    const int sr  = tid >> 2;           // row in tile
    const int scc = (tid & 3) * 8;      // k offset in tile (elements)
    const __bf16* srow = Xd + (size_t)(basem + sr) * D_IN + scc;

    // prologue: stage tile 0 into buffer 0
    if (tid < 128) {
        __builtin_prefetch(srow + 32, 0, 1);            // next tile -> global_prefetch_b8
        copy16_to_lds(srow, &As[0][sr * 32 + scc]);
    }
    async_wait_all();
    __syncthreads();

    for (int i = 0; i < 8; ++i) {          // K steps of 32 (K = 256)
        const int cur = i & 1;
        if (i < 7 && tid < 128) {          // stage next tile into the other buffer
            copy16_to_lds(srow + (i + 1) * 32, &As[cur ^ 1][sr * 32 + scc]);
        }
        const int kk = i * 32;

        // A fragment (16x32 bf16): lane l15 = row, lhi selects contiguous K-half (k = 16*lhi + s)
        union { v16bf v; u32x4 q[2]; } af;
        {
            int r  = m_sub * 16 + l15;
            int kb = lhi * 16;
            af.q[0] = *(const u32x4*)(&As[cur][r * 32 + kb]);        // K = kb .. kb+7
            af.q[1] = *(const u32x4*)(&As[cur][r * 32 + kb + 8]);    // K = kb+8 .. kb+15
        }
        // B fragments (32x16 bf16) straight from L2-hot Wt[n][k]: contiguous K
        union { v16bf v; u32x4 q[2]; } bf0, bf1;
        {
            int kb2 = lhi * 16;
            const __bf16* b0p = Wt + (size_t)(n0 + l15) * D_IN + kk + kb2;
            const __bf16* b1p = Wt + (size_t)(n0 + 16 + l15) * D_IN + kk + kb2;
            bf0.q[0] = *(const u32x4*)b0p;       bf0.q[1] = *(const u32x4*)(b0p + 8);
            bf1.q[0] = *(const u32x4*)b1p;       bf1.q[1] = *(const u32x4*)(b1p + 8);
        }
        acc0 = __builtin_amdgcn_wmma_f32_16x16x32_bf16(false, af.v, false, bf0.v,
                                                       (short)0, acc0, false, false);
        acc1 = __builtin_amdgcn_wmma_f32_16x16x32_bf16(false, af.v, false, bf1.v,
                                                       (short)0, acc1, false, false);

        async_wait_all();      // next tile landed in LDS (no-op for non-staging waves)
        __syncthreads();       // everyone done reading As[cur]; safe to overwrite next iter
    }

    // C/D layout: lane = n (0..15) | n (16..31 -> M+8); VGPR r -> row r (+8 for hi lanes)
    int m_base = basem + m_sub * 16 + lhi * 8;
    int ncol   = n0 + l15;
    for (int r = 0; r < 8; ++r) {
        H[(size_t)(m_base + r) * D_OUT + ncol]      = (__bf16)acc0[r];
        H[(size_t)(m_base + r) * D_OUT + ncol + 16] = (__bf16)acc1[r];
    }
}

// ---------- SpMM2: out[row] += a * h[col], one wave per nnz ----------
__global__ __launch_bounds__(256) void spmm_adj(const int* __restrict__ ar,
                                                const int* __restrict__ ac,
                                                const float* __restrict__ av,
                                                const __bf16* __restrict__ H,
                                                float* __restrict__ out, int nnz) {
    int e = blockIdx.x * 8 + (threadIdx.x >> 5);
    if (e >= nnz) return;
    int lane = threadIdx.x & 31;
    int row = ar[e];
    int col = ac[e];
    float a = av[e];
    const __bf16* hrow = H + (size_t)col * D_OUT + lane * 4;
    __builtin_prefetch(hrow, 0, 1);
    bf4 hv = *(const bf4*)hrow;
    float* o = out + (size_t)row * D_OUT + lane * 4;
    atomicAdd(o + 0, a * (float)hv[0]);
    atomicAdd(o + 1, a * (float)hv[1]);
    atomicAdd(o + 2, a * (float)hv[2]);
    atomicAdd(o + 3, a * (float)hv[3]);
}

// ---------- in-place ReLU ----------
__global__ void relu_k(float* __restrict__ out, int n4) {
    int i = blockIdx.x * blockDim.x + threadIdx.x;
    if (i >= n4) return;
    f32x4 v = ((f32x4*)out)[i];
    for (int r = 0; r < 4; ++r) v[r] = fmaxf(v[r], 0.0f);
    ((f32x4*)out)[i] = v;
}

extern "C" void kernel_launch(void* const* d_in, const int* in_sizes, int n_in,
                              void* d_out, int out_size, void* d_ws, size_t ws_size,
                              hipStream_t stream) {
    (void)in_sizes; (void)n_in; (void)out_size;
    if (ws_size < WS_BYTES) return;   // workspace too small: bail (output stays invalid)

    const int*   xr = (const int*)d_in[0];
    const int*   xc = (const int*)d_in[1];
    const float* xv = (const float*)d_in[2];
    const int*   ar = (const int*)d_in[3];
    const int*   ac = (const int*)d_in[4];
    const float* av = (const float*)d_in[5];
    const float* W  = (const float*)d_in[6];
    float*       out = (float*)d_out;

    __bf16* Xd = (__bf16*)d_ws;
    __bf16* Wt = Xd + WT_OFF;
    __bf16* H  = Xd + H_OFF;

    // 1) zero workspace (Xd must be 0 before scatter) and output accumulator
    zero_k<<<4096, 256, 0, stream>>>((u32x4*)d_ws, WS_BYTES / 16);
    zero_k<<<4096, 256, 0, stream>>>((u32x4*)d_out, (size_t)N_NODES * D_OUT * 4 / 16);
    // 2) W -> bf16, transposed
    prep_w<<<(D_IN * D_OUT + 255) / 256, 256, 0, stream>>>(W, Wt);
    // 3) dropout + densify X
    densify_x<<<(NNZX + 255) / 256, 256, 0, stream>>>(xr, xc, xv, Xd, NNZX);
    // 4) h = Xd @ W  (bf16 WMMA, f32 accumulate, async LDS pipeline)
    gemm_xw<<<N_NODES / 32, 256, 0, stream>>>(Xd, Wt, H);
    // 5) out += A @ h  (atomic scatter, L2-resident h and out)
    spmm_adj<<<NNZA / 8, 256, 0, stream>>>(ar, ac, av, H, out, NNZA);
    // 6) ReLU
    relu_k<<<(N_NODES * D_OUT / 4 + 255) / 256, 256, 0, stream>>>(out, N_NODES * D_OUT / 4);
}
